// GCN_16080357556338
// MI455X (gfx1250) — matrix operand order repaired and checked by reading, not compiled, not run on hardware
//
#include <hip/hip_runtime.h>
#include <hip/hip_bf16.h>

// GCN (TextGCN-like) on MI455X / gfx1250, wave32.
// GEMMs use v_wmma_f32_16x16x32_bf16 (fp32 accumulate); edge aggregation uses
// L2-resident global_atomic_add_f32 (h = 51MB < 192MB L2).

typedef __attribute__((ext_vector_type(16))) __bf16 v16bf;
typedef __attribute__((ext_vector_type(8)))  float  v8f;

#define G_NFEAT  512
#define G_NHID   128
#define G_NCLASS 20

// ---------------- degree / normalization ----------------

__global__ void k_deg(const int* __restrict__ col, const float* __restrict__ w,
                      float* __restrict__ deg, int E) {
  int t = blockIdx.x * blockDim.x + threadIdx.x;
  if (t < E) atomicAdd(&deg[col[t]], w[t]);
}

__global__ void k_dinv(const float* __restrict__ deg, float* __restrict__ dinv, int N) {
  int t = blockIdx.x * blockDim.x + threadIdx.x;
  if (t < N) {
    float d = deg[t] + 1.0f;        // +1 for self loop weight; always > 0
    dinv[t] = rsqrtf(d);
  }
}

// ---------------- GEMM1: h[N,128] = x[N,512] @ W1[512,128] (bf16 WMMA) ----------------
// One block = one 16-row strip of x staged to LDS as bf16; 8 waves cover the
// 8 N-tiles of 16. K loop: 512/32 = 16 WMMA per wave.

__global__ __launch_bounds__(256) void k_gemm1(const float* __restrict__ x,
                                               const float* __restrict__ W1,
                                               float* __restrict__ h) {
  __shared__ __bf16 As[16 * G_NFEAT];            // 16 KB
  const int m0 = blockIdx.x * 16;

  // cooperative stage + f32 -> bf16 convert (coalesced)
  for (int i = threadIdx.x; i < 16 * G_NFEAT; i += 256) {
    int r = i >> 9;                              // /512
    int c = i & (G_NFEAT - 1);
    As[i] = (__bf16)x[(size_t)(m0 + r) * G_NFEAT + c];
  }
  __syncthreads();

  const int wave = threadIdx.x >> 5;             // 0..7 -> N tile
  const int lane = threadIdx.x & 31;
  const int half = lane >> 4;
  const int l16  = lane & 15;
  const int n0   = wave * 16;

  v8f acc = {};
  for (int k0 = 0; k0 < G_NFEAT; k0 += 32) {
    // A fragment (16-bit A 16x32 layout): lane = M (mod 16), elems = K chunks
    v16bf a;
    const __bf16* arow = &As[l16 * G_NFEAT + k0];
#pragma unroll
    for (int i = 0; i < 8; ++i) a[i]     = arow[8 * half + i];
#pragma unroll
    for (int i = 0; i < 8; ++i) a[8 + i] = arow[16 + 8 * half + i];

    // B fragment: lane = N column (n0+l16), elems = K in same order as A
    v16bf b;
    const float* brow = W1 + (size_t)k0 * G_NHID + n0 + l16;
#pragma unroll
    for (int j = 0; j < 16; ++j) b[j] = (__bf16)brow[(size_t)(j < 8 ? 8 * half + j : 8 + 8 * half + j) * G_NHID];

    acc = __builtin_amdgcn_wmma_f32_16x16x32_bf16(false, a, false, b,
                                                  (short)0, acc, false, false);
  }

  // C/D layout: VGPR g -> M = g + 8*half, N = l16
#pragma unroll
  for (int g = 0; g < 8; ++g)
    h[(size_t)(m0 + g + 8 * half) * G_NHID + (n0 + l16)] = acc[g];
}

// ---------------- edge scatter, layer 1 (128 features) ----------------
// one wave per edge; 4 f32 atomics per lane, all L2-resident

__global__ __launch_bounds__(256) void k_scatter_hid(
    const int* __restrict__ row, const int* __restrict__ col,
    const float* __restrict__ w, const float* __restrict__ dinv,
    const float* __restrict__ h, float* __restrict__ agg, int E) {
  long long t = (long long)blockIdx.x * blockDim.x + threadIdx.x;
  int e = (int)(t >> 5);
  int lane = threadIdx.x & 31;
  if (e >= E) return;
  int r = row[e], c = col[e];
  float nrm = dinv[r] * w[e] * dinv[c];
  const float* hr = h + (size_t)r * G_NHID;
  float* oc = agg + (size_t)c * G_NHID;
#pragma unroll
  for (int i = 0; i < G_NHID / 32; ++i) {
    int f = lane + 32 * i;
    atomicAdd(&oc[f], hr[f] * nrm);
  }
}

// ---------------- self loop + bias + ReLU (fused, in-place on agg1) ----------------

__global__ __launch_bounds__(256) void k_self_relu(const float* __restrict__ h,
                                                   const float* __restrict__ dinv,
                                                   const float* __restrict__ b1,
                                                   float* __restrict__ agg,
                                                   long long total) {
  long long t = (long long)blockIdx.x * blockDim.x + threadIdx.x;
  if (t >= total) return;
  int i = (int)(t >> 7);                          // /128
  int f = (int)(t & (G_NHID - 1));
  float di = dinv[i];
  float v = agg[t] + h[t] * di * di + b1[f];
  agg[t] = v > 0.0f ? v : 0.0f;
}

// ---------------- GEMM2: h2[N,20] = h1[N,128] @ W2[128,20] (bf16 WMMA) ----------------
// 2 waves per block: N-tiles {0..15} and {16..31}, cols >= 20 masked to zero.

__global__ __launch_bounds__(64) void k_gemm2(const float* __restrict__ h1,
                                              const float* __restrict__ W2,
                                              float* __restrict__ h2) {
  __shared__ __bf16 As[16 * G_NHID];             // 4 KB
  const int m0 = blockIdx.x * 16;

  for (int i = threadIdx.x; i < 16 * G_NHID; i += 64) {
    int r = i >> 7;                              // /128
    int c = i & (G_NHID - 1);
    As[i] = (__bf16)h1[(size_t)(m0 + r) * G_NHID + c];
  }
  __syncthreads();

  const int wave = threadIdx.x >> 5;             // 0..1
  const int lane = threadIdx.x & 31;
  const int half = lane >> 4;
  const int l16  = lane & 15;
  const int n0   = wave * 16;

  v8f acc = {};
  for (int k0 = 0; k0 < G_NHID; k0 += 32) {
    v16bf a;
    const __bf16* arow = &As[l16 * G_NHID + k0];
#pragma unroll
    for (int i = 0; i < 8; ++i) a[i]     = arow[8 * half + i];
#pragma unroll
    for (int i = 0; i < 8; ++i) a[8 + i] = arow[16 + 8 * half + i];

    v16bf b;
    const float* brow = W2 + (size_t)k0 * G_NCLASS + n0 + l16;
#pragma unroll
    for (int j = 0; j < 16; ++j)
      b[j] = (n0 + l16 < G_NCLASS) ? (__bf16)brow[(size_t)(j < 8 ? 8 * half + j : 8 + 8 * half + j) * G_NCLASS] : (__bf16)0.0f;

    acc = __builtin_amdgcn_wmma_f32_16x16x32_bf16(false, a, false, b,
                                                  (short)0, acc, false, false);
  }

  int ncol = n0 + l16;
  if (ncol < G_NCLASS) {
#pragma unroll
    for (int g = 0; g < 8; ++g)
      h2[(size_t)(m0 + g + 8 * half) * G_NCLASS + ncol] = acc[g];
  }
}

// ---------------- edge scatter, layer 2 (20 features) ----------------

__global__ __launch_bounds__(256) void k_scatter_cls(
    const int* __restrict__ row, const int* __restrict__ col,
    const float* __restrict__ w, const float* __restrict__ dinv,
    const float* __restrict__ h2, float* __restrict__ agg2, int E) {
  long long t = (long long)blockIdx.x * blockDim.x + threadIdx.x;
  int e = (int)(t >> 5);
  int lane = threadIdx.x & 31;
  if (e >= E) return;
  int r = row[e], c = col[e];
  float nrm = dinv[r] * w[e] * dinv[c];
  if (lane < G_NCLASS)
    atomicAdd(&agg2[(size_t)c * G_NCLASS + lane],
              h2[(size_t)r * G_NCLASS + lane] * nrm);
}

// ---------------- self loop + bias + log_softmax (one wave32 per node) ----------------

__global__ __launch_bounds__(256) void k_finalize(const float* __restrict__ h2,
                                                  const float* __restrict__ agg2,
                                                  const float* __restrict__ dinv,
                                                  const float* __restrict__ b2,
                                                  float* __restrict__ out, int N) {
  long long t = (long long)blockIdx.x * blockDim.x + threadIdx.x;
  int i = (int)(t >> 5);
  int lane = threadIdx.x & 31;
  if (i >= N) return;
  float di = dinv[i];
  float di2 = di * di;
  float val = 0.0f;
  float m = -__builtin_inff();
  if (lane < G_NCLASS) {
    val = agg2[(size_t)i * G_NCLASS + lane] +
          h2[(size_t)i * G_NCLASS + lane] * di2 + b2[lane];
    m = val;
  }
#pragma unroll
  for (int s = 16; s > 0; s >>= 1) m = fmaxf(m, __shfl_xor(m, s, 32));
  float ex = (lane < G_NCLASS) ? __expf(val - m) : 0.0f;
#pragma unroll
  for (int s = 16; s > 0; s >>= 1) ex += __shfl_xor(ex, s, 32);
  if (lane < G_NCLASS)
    out[(size_t)i * G_NCLASS + lane] = val - m - __logf(ex);
}

// ---------------- launcher ----------------

extern "C" void kernel_launch(void* const* d_in, const int* in_sizes, int n_in,
                              void* d_out, int out_size, void* d_ws, size_t ws_size,
                              hipStream_t stream) {
  const float* x  = (const float*)d_in[0];
  const int*   ei = (const int*)d_in[1];
  const float* ew = (const float*)d_in[2];
  const float* W1 = (const float*)d_in[3];
  const float* b1 = (const float*)d_in[4];
  const float* W2 = (const float*)d_in[5];
  const float* b2 = (const float*)d_in[6];
  float* out = (float*)d_out;

  const int E = in_sizes[2];
  const int N = in_sizes[0] / G_NFEAT;     // 100000, multiple of 16
  const int* row = ei;                     // edge_index[0]
  const int* col = ei + E;                 // edge_index[1]

  // workspace layout (~104 MB): deg | dinv | h (N*128) | agg1 (N*128)
  // h2 / agg2 alias the h region (h is dead after k_self_relu)
  float* deg  = (float*)d_ws;
  float* dinv = deg + N;
  float* h    = dinv + N;
  float* agg1 = h + (size_t)N * G_NHID;
  float* h2   = h;                                  // N*20
  float* agg2 = h + (size_t)N * G_NCLASS;           // N*20, disjoint from h2

  hipMemsetAsync(deg, 0, (size_t)N * sizeof(float), stream);
  hipMemsetAsync(agg1, 0, (size_t)N * G_NHID * sizeof(float), stream);

  k_deg<<<(E + 255) / 256, 256, 0, stream>>>(col, ew, deg, E);
  k_dinv<<<(N + 255) / 256, 256, 0, stream>>>(deg, dinv, N);

  k_gemm1<<<N / 16, 256, 0, stream>>>(x, W1, h);

  long long sth = (long long)E * 32;
  k_scatter_hid<<<(unsigned)((sth + 255) / 256), 256, 0, stream>>>(
      row, col, ew, dinv, h, agg1, E);

  long long tot = (long long)N * G_NHID;
  k_self_relu<<<(unsigned)((tot + 255) / 256), 256, 0, stream>>>(
      h, dinv, b1, agg1, tot);

  k_gemm2<<<N / 16, 64, 0, stream>>>(agg1, W2, h2);

  hipMemsetAsync(agg2, 0, (size_t)N * G_NCLASS * sizeof(float), stream);
  k_scatter_cls<<<(unsigned)((sth + 255) / 256), 256, 0, stream>>>(
      row, col, ew, dinv, h2, agg2, E);

  long long fth = (long long)N * 32;
  k_finalize<<<(unsigned)((fth + 255) / 256), 256, 0, stream>>>(
      h2, agg2, dinv, b2, out, N);
}
